// SnapML_Attention_34308198760844
// MI455X (gfx1250) — matrix operand order repaired and checked by reading, not compiled, not run on hardware
//
#include <hip/hip_runtime.h>
#include <hip/hip_bf16.h>

// ---------------------------------------------------------------------------
// LeViT attention on MI455X (gfx1250): all GEMMs via v_wmma_f32_16x16x32_f16.
//   k_cvt  : f32 -> f16 for x and all weights
//   k_qkv  : [2304x384] x {qw,kw,vw}^T -> q(h,N,32) k(h,N,32) vT(h,128,N)
//            2x4 register-blocked (8 WMMA accumulators / wave)
//   k_attn : per (head, 16-row tile): S=QK^T (WMMA) + analytic bias
//            -> softmax in LDS -> out = P V (72 chained WMMAs / wave)
//   k_proj : [2304x1024] x proj_w^T + b -> d_out (f32), 2x4 register-blocked
// bias_idxs input is never read: idx == |y1-y2|*48 + |x1-x2| is recomputed.
// ---------------------------------------------------------------------------

typedef __attribute__((ext_vector_type(16))) _Float16 v16h;
typedef __attribute__((ext_vector_type(8)))  _Float16 v8h;
typedef __attribute__((ext_vector_type(8)))  float    v8f;

#define N_TOK   2304
#define DIM_    384
#define KD      32
#define NH      8
#define DV      128
#define DH      1024     // NH*DV
#define RES     48
#define SCALE_F 0.17677669529663687f   // 1/sqrt(32)

#define MT_TILES   144   // N_TOK/16
#define QKV_OT     96    // (256+256+1024)/16
#define PROJ_OT    24    // 384/16

// Dynamic LDS layout for k_attn: bias[2304]f32 | rowsum[16]f32(+pad) | P[16*2304]f16
#define SMEM_BIAS_B   (N_TOK * 4)
#define SMEM_RS_B     64
#define SMEM_BYTES    (SMEM_BIAS_B + SMEM_RS_B + 16 * N_TOK * 2)

// ---- WMMA operand loaders (contiguous k = 16*half + s, same order for A and B) -------
// A operand (16x32 f16): lane L holds row (L&15); K = 16*(L>>4)+{0..15}
// packed sequentially.  rowk0 = &M[row][k0] (row-major).
__device__ __forceinline__ v16h load_opA(const _Float16* rowk0, int half) {
  const _Float16* p = rowk0 + half * 16;
  v8h lo = *(const v8h*)(p);
  v8h hi = *(const v8h*)(p + 8);
  v16h a;
#pragma unroll
  for (int t = 0; t < 8; ++t) { a[t] = lo[t]; a[t + 8] = hi[t]; }
  return a;
}

// B operand (32x16 f16): lane L holds column (L&15); K = 16*(L>>4)+{0..15}
// packed sequentially.  rowk0 = &Bt[col][k0] where Bt is B transposed
// (i.e. a weight stored (out,in), or k / vT rows), row-major.
__device__ __forceinline__ v16h load_opB(const _Float16* rowk0, int half) {
  const _Float16* p = rowk0 + half * 16;
  v8h lo = *(const v8h*)(p);
  v8h hi = *(const v8h*)(p + 8);
  v16h b;
#pragma unroll
  for (int t = 0; t < 8; ++t) { b[t] = lo[t]; b[t + 8] = hi[t]; }
  return b;
}

__device__ __forceinline__ v8f wmma16(v16h a, v16h b, v8f c) {
  return __builtin_amdgcn_wmma_f32_16x16x32_f16(
      /*neg_a=*/false, a, /*neg_b=*/false, b,
      /*c_mod=*/(short)0, c, /*reuse_a=*/false, /*reuse_b=*/false);
}

// ---------------------------------------------------------------------------
__global__ void k_cvt(const float* __restrict__ s, _Float16* __restrict__ d, int n) {
  int i = blockIdx.x * blockDim.x + threadIdx.x;
  if (i < n) d[i] = (_Float16)s[i];
}

// ---------------------------------------------------------------------------
// QKV projection, 2 m-tiles x 4 o-tiles per wave (1.5 b128 loads per WMMA).
// 72 m-groups x 24 o-groups = 1728 waves. o-groups 0..3 -> q, 4..7 -> k,
// 8..23 -> v (stored transposed per head).
__global__ void __launch_bounds__(256)
k_qkv(const _Float16* __restrict__ xh,
      const _Float16* __restrict__ qw, const _Float16* __restrict__ kw,
      const _Float16* __restrict__ vw,
      const float* __restrict__ qb, const float* __restrict__ kb,
      const float* __restrict__ vb,
      _Float16* __restrict__ qh, _Float16* __restrict__ kh,
      _Float16* __restrict__ vT) {
  const int wave = (blockIdx.x * blockDim.x + threadIdx.x) >> 5;
  const int mg   = wave / 24;            // 0..71  : rows mg*32 .. +31
  const int og   = wave % 24;            // 0..23  : o-tiles og*4 .. +3 (uniform)
  const int lane = threadIdx.x & 31;
  const int r    = lane & 15;
  const int half = lane >> 4;
  const int m0   = mg * 32;
  const int ot0  = og * 4;

  const _Float16* W; const float* bia; int obase;
  if (ot0 < 16)      { W = qw; bia = qb; obase = ot0 * 16; }
  else if (ot0 < 32) { W = kw; bia = kb; obase = (ot0 - 16) * 16; }
  else               { W = vw; bia = vb; obase = (ot0 - 32) * 16; }

  const _Float16* arow[2];
  arow[0] = xh + (size_t)(m0 + r) * DIM_;
  arow[1] = xh + (size_t)(m0 + 16 + r) * DIM_;
  const _Float16* brow[4];
#pragma unroll
  for (int j = 0; j < 4; ++j)
    brow[j] = W + (size_t)(obase + j * 16 + r) * DIM_;

  v8f acc[2][4] = {};
  for (int k = 0; k < DIM_; k += 32) {
    v16h a0 = load_opA(arow[0] + k, half);
    v16h a1 = load_opA(arow[1] + k, half);
#pragma unroll
    for (int j = 0; j < 4; ++j) {
      v16h b = load_opB(brow[j] + k, half);
      acc[0][j] = wmma16(a0, b, acc[0][j]);
      acc[1][j] = wmma16(a1, b, acc[1][j]);
    }
  }

#pragma unroll
  for (int j = 0; j < 4; ++j) {
    const int olocal = obase + j * 16 + r;
    const float bv = bia[olocal];
    if (ot0 < 32) {
      _Float16* dst = (ot0 < 16) ? qh : kh;
      const int hh = olocal >> 5, kd = olocal & 31;
#pragma unroll
      for (int mi = 0; mi < 2; ++mi)
#pragma unroll
        for (int i = 0; i < 8; ++i) {
          const int mgl = m0 + mi * 16 + half * 8 + i;
          dst[((size_t)hh * N_TOK + mgl) * KD + kd] = (_Float16)(acc[mi][j][i] + bv);
        }
    } else {
      const int hh = olocal >> 7, dv = olocal & 127;
#pragma unroll
      for (int mi = 0; mi < 2; ++mi) {
        v8h pack;
#pragma unroll
        for (int i = 0; i < 8; ++i) pack[i] = (_Float16)(acc[mi][j][i] + bv);
        // D-layout rows are 8 consecutive tokens -> one contiguous 16B store.
        *(v8h*)(vT + ((size_t)hh * DV + dv) * N_TOK + m0 + mi * 16 + half * 8) = pack;
      }
    }
  }
}

// ---------------------------------------------------------------------------
// Attention core: one block = (head, 16-row tile). 8 waves, 83KB dynamic LDS.
__global__ void __launch_bounds__(256)
k_attn(const _Float16* __restrict__ qh, const _Float16* __restrict__ kh,
       const _Float16* __restrict__ vT, const float* __restrict__ ab,
       _Float16* __restrict__ outbuf) {
  extern __shared__ char smem[];
  float*    biasS  = (float*)smem;                                // [2304]
  float*    rowsum = (float*)(smem + SMEM_BIAS_B);                // [16]
  _Float16* P      = (_Float16*)(smem + SMEM_BIAS_B + SMEM_RS_B); // [16][2304]

  const int h  = blockIdx.x / MT_TILES;
  const int mt = blockIdx.x % MT_TILES;
  const int m0 = mt * 16;
  const int wv = threadIdx.x >> 5;
  const int lane = threadIdx.x & 31;
  const int r    = lane & 15;
  const int half = lane >> 4;

  for (int i = threadIdx.x; i < N_TOK; i += 256) biasS[i] = ab[h * N_TOK + i];
  __syncthreads();

  // Per-lane (y,x) coords of the 8 rows this lane owns in the D layout.
  int my[8], mxx[8];
#pragma unroll
  for (int i = 0; i < 8; ++i) {
    const int mg = m0 + half * 8 + i;
    my[i] = mg / RES; mxx[i] = mg - my[i] * RES;
  }

  // ---- Phase 1: S = (Q K^T)*scale + bias(analytic) -> LDS (f16) ----------
  const v16h aQ = load_opA(qh + ((size_t)h * N_TOK + m0 + r) * KD, half);
  for (int j = wv; j < MT_TILES; j += 8) {
    const v16h bK = load_opB(kh + ((size_t)h * N_TOK + j * 16 + r) * KD, half);
    v8f z = {};
    v8f s = wmma16(aQ, bK, z);
    const int ng = j * 16 + r;
    const int y2 = ng / RES, x2 = ng - (ng / RES) * RES;
#pragma unroll
    for (int i = 0; i < 8; ++i) {
      int dy = my[i] - y2;  dy = dy < 0 ? -dy : dy;
      int dx = mxx[i] - x2; dx = dx < 0 ? -dx : dx;
      const float val = s[i] * SCALE_F + biasS[dy * RES + dx];
      P[(half * 8 + i) * N_TOK + ng] = (_Float16)val;
    }
  }
  __syncthreads();

  // ---- Phase 2: softmax per row (2 rows per wave, lane-parallel cols) ----
  for (int q2 = 0; q2 < 2; ++q2) {
    const int rr = wv * 2 + q2;
    _Float16* prow = P + rr * N_TOK;
    float mval = -3.0e38f;
    for (int c = lane; c < N_TOK; c += 32) mval = fmaxf(mval, (float)prow[c]);
#pragma unroll
    for (int off = 16; off >= 1; off >>= 1)
      mval = fmaxf(mval, __shfl_xor(mval, off, 32));
    float ssum = 0.f;
    for (int c = lane; c < N_TOK; c += 32) {
      const float e = __expf((float)prow[c] - mval);
      ssum += e;
      prow[c] = (_Float16)e;     // unnormalized; 1/sum folded into PV epilogue
    }
#pragma unroll
    for (int off = 16; off >= 1; off >>= 1) ssum += __shfl_xor(ssum, off, 32);
    if (lane == 0) rowsum[rr] = ssum;
  }
  __syncthreads();

  // ---- Phase 3: out = P V, one dv-tile per wave, 72 chained WMMAs --------
  float inv[8];
#pragma unroll
  for (int i = 0; i < 8; ++i) inv[i] = 1.0f / rowsum[half * 8 + i];

  v8f acc = {};
  const int dv0 = wv * 16;
  const _Float16* vrow = vT + ((size_t)h * DV + dv0 + r) * N_TOK;
  const _Float16* prow = P + r * N_TOK;
  for (int t = 0; t < N_TOK; t += 32)
    acc = wmma16(load_opA(prow + t, half), load_opB(vrow + t, half), acc);

#pragma unroll
  for (int i = 0; i < 8; ++i) {
    const int mg = m0 + half * 8 + i;
    outbuf[(size_t)mg * DH + h * DV + dv0 + r] = (_Float16)(acc[i] * inv[i]);
  }
}

// ---------------------------------------------------------------------------
// Final projection: y = out @ proj_w^T + proj_b (f32). 2x4 register-blocked:
// 72 m-groups x 6 o-groups = 432 waves.
__global__ void __launch_bounds__(256)
k_proj(const _Float16* __restrict__ ob, const _Float16* __restrict__ pw,
       const float* __restrict__ pb, float* __restrict__ y) {
  const int wave = (blockIdx.x * blockDim.x + threadIdx.x) >> 5;
  const int mg   = wave / 6;             // 0..71
  const int og   = wave % 6;             // 0..5
  const int lane = threadIdx.x & 31;
  const int r    = lane & 15;
  const int half = lane >> 4;
  const int m0   = mg * 32;
  const int o0   = og * 64;

  const _Float16* arow[2];
  arow[0] = ob + (size_t)(m0 + r) * DH;
  arow[1] = ob + (size_t)(m0 + 16 + r) * DH;
  const _Float16* brow[4];
#pragma unroll
  for (int j = 0; j < 4; ++j)
    brow[j] = pw + (size_t)(o0 + j * 16 + r) * DH;

  v8f acc[2][4] = {};
  for (int k = 0; k < DH; k += 32) {
    v16h a0 = load_opA(arow[0] + k, half);
    v16h a1 = load_opA(arow[1] + k, half);
#pragma unroll
    for (int j = 0; j < 4; ++j) {
      v16h b = load_opB(brow[j] + k, half);
      acc[0][j] = wmma16(a0, b, acc[0][j]);
      acc[1][j] = wmma16(a1, b, acc[1][j]);
    }
  }

#pragma unroll
  for (int j = 0; j < 4; ++j) {
    const int o = o0 + j * 16 + r;
    const float bv = pb[o];
#pragma unroll
    for (int mi = 0; mi < 2; ++mi)
#pragma unroll
      for (int i = 0; i < 8; ++i)
        y[(size_t)(m0 + mi * 16 + half * 8 + i) * DIM_ + o] = acc[mi][j][i] + bv;
  }
}

// ---------------------------------------------------------------------------
extern "C" void kernel_launch(void* const* d_in, const int* in_sizes, int n_in,
                              void* d_out, int out_size, void* d_ws, size_t ws_size,
                              hipStream_t stream) {
  (void)in_sizes; (void)n_in; (void)out_size; (void)ws_size;
  const float* x  = (const float*)d_in[0];
  const float* qw = (const float*)d_in[1];
  const float* qb = (const float*)d_in[2];
  const float* kw = (const float*)d_in[3];
  const float* kb = (const float*)d_in[4];
  const float* vw = (const float*)d_in[5];
  const float* vb = (const float*)d_in[6];
  const float* pw = (const float*)d_in[7];
  const float* pb = (const float*)d_in[8];
  const float* ab = (const float*)d_in[9];
  // d_in[10] = bias_idxs : recomputed analytically on-chip, never read.
  float* y = (float*)d_out;

  char* ws = (char*)d_ws;
  size_t off = 0;
  auto alloc = [&](size_t bytes) -> char* {
    char* p = ws + off; off += (bytes + 255) & ~(size_t)255; return p;
  };
  _Float16* xh  = (_Float16*)alloc((size_t)N_TOK * DIM_ * 2);
  _Float16* qwh = (_Float16*)alloc((size_t)256 * DIM_ * 2);
  _Float16* kwh = (_Float16*)alloc((size_t)256 * DIM_ * 2);
  _Float16* vwh = (_Float16*)alloc((size_t)DH * DIM_ * 2);
  _Float16* pwh = (_Float16*)alloc((size_t)DIM_ * DH * 2);
  _Float16* qhB = (_Float16*)alloc((size_t)NH * N_TOK * KD * 2);
  _Float16* khB = (_Float16*)alloc((size_t)NH * N_TOK * KD * 2);
  _Float16* vTB = (_Float16*)alloc((size_t)NH * DV * N_TOK * 2);
  _Float16* obB = (_Float16*)alloc((size_t)N_TOK * DH * 2);

  auto cvt = [&](const float* s, _Float16* d, int n) {
    k_cvt<<<(n + 255) / 256, 256, 0, stream>>>(s, d, n);
  };
  cvt(x,  xh,  N_TOK * DIM_);
  cvt(qw, qwh, 256 * DIM_);
  cvt(kw, kwh, 256 * DIM_);
  cvt(vw, vwh, DH * DIM_);
  cvt(pw, pwh, DIM_ * DH);

  // 72*24 = 1728 waves, 8 waves/block
  k_qkv<<<1728 / 8, 256, 0, stream>>>(
      xh, qwh, kwh, vwh, qb, kb, vb, qhB, khB, vTB);

  static_assert(SMEM_BYTES == 83008, "lds layout");
  (void)hipFuncSetAttribute(reinterpret_cast<const void*>(k_attn),
                            hipFuncAttributeMaxDynamicSharedMemorySize,
                            SMEM_BYTES);
  k_attn<<<NH * MT_TILES, 256, SMEM_BYTES, stream>>>(qhB, khB, vTB, ab, obB);

  // 72*6 = 432 waves, 8 waves/block
  k_proj<<<432 / 8, 256, 0, stream>>>(obB, pwh, pb, y);
}
